// AdapterLayer_63221918597770
// MI455X (gfx1250) — matrix operand order repaired and checked by reading, not compiled, not run on hardware
//
#include <hip/hip_runtime.h>

typedef __attribute__((ext_vector_type(16))) _Float16 v16h;
typedef __attribute__((ext_vector_type(8)))  _Float16 v8h;
typedef __attribute__((ext_vector_type(8)))  float    v8f;

#define HW   16384      // 128*128
#define IMW  128
#define NB   8
#define NEXP 4

constexpr int GM_PLAIN   = 0;  // Y(f16) = W@X
constexpr int GM_SILU    = 1;  // Y(f16) = silu(W@X)
constexpr int GM_BIASMUL = 2;  // Y(f16) = (W@X + bias) * mul
constexpr int GM_ACCUM   = 3;  // Yf32  += gate * (W@X)

// ---------------------------------------------------------------------------
// 1x1-conv GEMM: per batch b, Y[m][n] = op( sum_k W[m][k] * X[k][n] ).
// Block = 256 threads (8 waves), tile = 128 pixels, each wave one 16-col tile,
// loops over all M tiles reusing its B fragments (held in VGPRs; M, K are
// compile-time so no dynamically-indexed locals -> no scratch spills).
// WMMA f32_16x16x32_f16, f32 accumulate.
// ---------------------------------------------------------------------------
template <typename TIN, int MODE, int M, int K>
__global__ __launch_bounds__(256)
void gemm_1x1(const TIN* __restrict__ X, const float* __restrict__ W,
              const float* __restrict__ bias, const _Float16* __restrict__ mul,
              void* __restrict__ Yout, int xbs, int ybs,
              const float* __restrict__ gates, int e)
{
    const int b = blockIdx.y;
    const float gate = gates[b * NEXP + e];
    if (gate == 0.0f) return;

    constexpr int K32 = (K + 31) & ~31;      // zero-pad K to WMMA depth
    constexpr int KP  = K32 + 8;             // padded LDS row stride (16B mult.)
    constexpr int NKC = K32 >> 5;            // 1 or 2 K-chunks
    constexpr int MTS = M >> 4;              // M tiles
    __shared__ alignas(16) _Float16 lw[M * KP];
    __shared__ alignas(16) _Float16 lx[128 * KP];
    __shared__ float lb[M];

    const int t = threadIdx.x;
    // stage weights (f32 -> f16, zero-pad K..K32)
    for (int idx = t; idx < M * K32; idx += 256) {
        int m = idx / K32, k = idx - m * K32;
        lw[m * KP + k] = (k < K) ? (_Float16)W[m * K + k] : (_Float16)0.0f;
    }
    if (MODE == GM_BIASMUL) for (int m = t; m < M; m += 256) lb[m] = bias[m];

    // stage activations transposed: lx[pixel][k]
    const TIN* Xb = X + (size_t)b * xbs;
    const int p0 = blockIdx.x * 128;
    for (int idx = t; idx < 128 * K32; idx += 256) {
        int k = idx >> 7, n = idx & 127;
        float v = (k < K) ? (float)Xb[(size_t)k * HW + p0 + n] : 0.0f;
        lx[n * KP + k] = (_Float16)v;
    }
    if (t < 32) __builtin_prefetch(Xb + (size_t)t * HW + p0 + 128, 0, 0);
    __syncthreads();

    const int wv = t >> 5, lane = t & 31;
    const int lhi = lane >> 4, l15 = lane & 15;

    union Frag { v16h v; v8h h[2]; };
    // B frags: lane n -> col n, 16 consecutive k per half-wave
    Frag bf0, bf1;
    {
        const _Float16* src = &lx[(wv * 16 + l15) * KP + lhi * 16];
        bf0.h[0] = *(const v8h*)(src);
        bf0.h[1] = *(const v8h*)(src + 8);
        if constexpr (NKC == 2) {
            bf1.h[0] = *(const v8h*)(src + 32);
            bf1.h[1] = *(const v8h*)(src + 40);
        }
    }

    const int nglob = p0 + wv * 16 + l15;
    for (int mt = 0; mt < MTS; ++mt) {
        v8f acc;
        for (int g = 0; g < 8; ++g)
            acc[g] = (MODE == GM_BIASMUL) ? lb[mt * 16 + g + lhi * 8] : 0.0f;
        // A frags: lane l -> row l15, 16 consecutive k per half-wave
        const _Float16* wsrc = &lw[(mt * 16 + l15) * KP + lhi * 16];
        Frag af0;
        af0.h[0] = *(const v8h*)(wsrc);
        af0.h[1] = *(const v8h*)(wsrc + 8);
        acc = __builtin_amdgcn_wmma_f32_16x16x32_f16(
            false, af0.v, false, bf0.v, (short)0, acc, false, false);
        if constexpr (NKC == 2) {
            Frag af1;
            af1.h[0] = *(const v8h*)(wsrc + 32);
            af1.h[1] = *(const v8h*)(wsrc + 40);
            acc = __builtin_amdgcn_wmma_f32_16x16x32_f16(
                false, af1.v, false, bf1.v, (short)0, acc, false, false);
        }
        if (MODE == GM_ACCUM) {
            float* Y = (float*)Yout + (size_t)b * ybs;
            for (int g = 0; g < 8; ++g) {
                int m = mt * 16 + g + lhi * 8;
                Y[(size_t)m * HW + nglob] += gate * acc[g];
            }
        } else {
            _Float16* Y = (_Float16*)Yout + (size_t)b * ybs;
            for (int g = 0; g < 8; ++g) {
                int m = mt * 16 + g + lhi * 8;
                float v = acc[g];
                if (MODE == GM_SILU) v = v / (1.0f + __expf(-v));
                if (MODE == GM_BIASMUL)
                    v *= (float)mul[(size_t)b * ybs + (size_t)m * HW + nglob];
                Y[(size_t)m * HW + nglob] = (_Float16)v;
            }
        }
    }
}

// ---------------------------------------------------------------------------
// Depthwise KSxKS conv, zero pad, f16 in/out, f32 weights+bias.
// ---------------------------------------------------------------------------
template <int KS>
__global__ __launch_bounds__(256)
void dwconv(const _Float16* __restrict__ in, const float* __restrict__ w,
            const float* __restrict__ bias, _Float16* __restrict__ out,
            int C, const float* __restrict__ gates, int e)
{
    int idx = blockIdx.x * 256 + threadIdx.x;   // b*C*HW + c*HW + hw
    int chw = C * HW;
    int b = idx / chw;
    if (gates[b * NEXP + e] == 0.0f) return;    // uniform per block (chw % 256 == 0)
    int rem = idx - b * chw;
    int c = rem >> 14, hw = rem & (HW - 1);
    int y = hw >> 7, x = hw & 127;
    const _Float16* ip = in + (size_t)b * chw + (size_t)c * HW;
    const float* wp = w + c * KS * KS;
    const int P = KS / 2;
    float s = bias[c];
    for (int dy = 0; dy < KS; ++dy) {
        int yy = y + dy - P; if (yy < 0 || yy >= IMW) continue;
        for (int dx = 0; dx < KS; ++dx) {
            int xx = x + dx - P; if (xx < 0 || xx >= IMW) continue;
            s += wp[dy * KS + dx] * (float)ip[yy * IMW + xx];
        }
    }
    out[idx] = (_Float16)s;
}

// ---------------------------------------------------------------------------
// Per-patch 8x8 circular convolution: o = ifft2(fft2(q)*fft2(k)).real
// 4 (channel,patch) units per 256-thread block, staged in LDS.
// ---------------------------------------------------------------------------
__global__ __launch_bounds__(256)
void circconv(const _Float16* __restrict__ q, const _Float16* __restrict__ k,
              _Float16* __restrict__ o, int r, int kbs,
              const float* __restrict__ gates, int e)
{
    __shared__ float lq[4][64];
    __shared__ float lk[4][64];
    int t = threadIdx.x;
    int u = t >> 6, pix = t & 63;
    int unit = blockIdx.x * 4 + u;
    int pc = r * 256;                        // units per batch (divisible by 4)
    int b = unit / pc;
    if (gates[b * NEXP + e] == 0.0f) return;
    int rem = unit - b * pc;
    int c = rem >> 8, patch = rem & 255;
    int ph = patch >> 4, pw = patch & 15;
    int i = pix >> 3, j = pix & 7;
    size_t qbase = (size_t)b * r * HW + (size_t)c * HW + (ph * 8) * IMW + pw * 8;
    size_t kbase = (size_t)b * kbs    + (size_t)c * HW + (ph * 8) * IMW + pw * 8;
    lq[u][pix] = (float)q[qbase + i * IMW + j];
    lk[u][pix] = (float)k[kbase + i * IMW + j];
    __syncthreads();
    float s = 0.0f;
    for (int sv = 0; sv < 64; ++sv) {
        int si = sv >> 3, sj = sv & 7;
        s += lq[u][sv] * lk[u][((i - si) & 7) * 8 + ((j - sj) & 7)];
    }
    o[qbase + i * IMW + j] = (_Float16)s;
}

// ---------------------------------------------------------------------------
// Channel LayerNorm (WithBias) at each pixel, then multiply by v.
// ---------------------------------------------------------------------------
__global__ __launch_bounds__(256)
void ln_mul(const _Float16* __restrict__ o, const _Float16* __restrict__ v,
            const float* __restrict__ lnw, const float* __restrict__ lnb,
            _Float16* __restrict__ out, int r, int vbs, int vco,
            const float* __restrict__ gates, int e)
{
    int b = blockIdx.y;
    if (gates[b * NEXP + e] == 0.0f) return;
    __shared__ _Float16 lo[64 * 256];
    int t = threadIdx.x;
    int p0 = blockIdx.x * 256;
    size_t ob = (size_t)b * r * HW + p0;
    for (int c = 0; c < r; ++c) lo[c * 256 + t] = o[ob + (size_t)c * HW + t];
    __syncthreads();
    float sum = 0.0f, sq = 0.0f;
    for (int c = 0; c < r; ++c) {
        float x = (float)lo[c * 256 + t];
        sum += x; sq += x * x;
    }
    float mu = sum / r;
    float rs = rsqrtf(sq / r - mu * mu + 1e-5f);
    size_t vb = (size_t)b * vbs + (size_t)vco * HW + p0;
    for (int c = 0; c < r; ++c) {
        float x = ((float)lo[c * 256 + t] - mu) * rs * lnw[c] + lnb[c];
        float vv = (float)v[vb + (size_t)c * HW + t];
        out[ob + (size_t)c * HW + t] = (_Float16)(x * vv);
    }
}

// ---------------------------------------------------------------------------
// Gating: adaptive avg-pool, logits, noisy softmax, top-2 gates.
// ---------------------------------------------------------------------------
__global__ __launch_bounds__(256)
void pool_kernel(const float* __restrict__ x, float* __restrict__ pooled)
{
    __shared__ float s[256];
    int bc = blockIdx.x, t = threadIdx.x;
    const float* p = x + (size_t)bc * HW;
    float sum = 0.0f;
    for (int i = t; i < HW; i += 256) sum += p[i];
    s[t] = sum; __syncthreads();
    for (int st = 128; st > 0; st >>= 1) {
        if (t < st) s[t] += s[t + st];
        __syncthreads();
    }
    if (t == 0) pooled[bc] = s[0] * (1.0f / HW);
}

__global__ void gate_kernel(const float* __restrict__ pooled,
                            const float* __restrict__ freq,
                            const float* __restrict__ noise,
                            const float* __restrict__ wg,
                            const float* __restrict__ wf,
                            float* __restrict__ gates, float* __restrict__ gsum)
{
    int b = threadIdx.x;
    if (b >= NB) return;
    float lg[NEXP];
    for (int e = 0; e < NEXP; ++e) {
        float s = 0.0f;
        for (int c = 0; c < 64; ++c)
            s += pooled[b * 64 + c] * wg[e * 64 + c] + freq[b * 64 + c] * wf[e * 64 + c];
        lg[e] = s + noise[b * NEXP + e] * (1.0f / NEXP);
    }
    float mx = lg[0];
    for (int e = 1; e < NEXP; ++e) mx = fmaxf(mx, lg[e]);
    float den = 0.0f, sc[NEXP];
    for (int e = 0; e < NEXP; ++e) { sc[e] = __expf(lg[e] - mx); den += sc[e]; }
    for (int e = 0; e < NEXP; ++e) sc[e] /= den;
    int i1 = 0;
    for (int e = 1; e < NEXP; ++e) if (sc[e] > sc[i1]) i1 = e;   // first max (jax tie rule)
    int i2 = -1;
    for (int e = 0; e < NEXP; ++e)
        if (e != i1 && (i2 < 0 || sc[e] > sc[i2])) i2 = e;
    for (int e = 0; e < NEXP; ++e)
        gates[b * NEXP + e] = (e == i1 || e == i2) ? sc[e] : 0.0f;
    gsum[b] = sc[i1] + sc[i2];
}

__global__ __launch_bounds__(256)
void init_out(const float* __restrict__ x, const float* __restrict__ gsum,
              float* __restrict__ out)
{
    size_t i = (size_t)blockIdx.x * 256 + threadIdx.x;  // 8*64*16384 total
    int b = (int)(i >> 20);                             // 64*16384 = 2^20
    out[i] = gsum[b] * x[i];
}

// ---------------------------------------------------------------------------
// One expert, rank RK known at compile time -> fully static WMMA shapes.
// ---------------------------------------------------------------------------
template <int RK>
static void expert_pipeline(void* const* d_in, int e,
                            const float* x, const float* shr, float* out,
                            float* gates, char* R0, char* R1, char* R2,
                            char* R3, char* R4, hipStream_t stream)
{
    constexpr int r = RK;
    const size_t MB = 1024 * 1024;
    const float* p0w  = (const float*)d_in[6 + e * 13 + 0];
    const float* p1w  = (const float*)d_in[6 + e * 13 + 1];
    const float* p2w  = (const float*)d_in[6 + e * 13 + 2];
    const float* qw   = (const float*)d_in[6 + e * 13 + 3];
    const float* qdw  = (const float*)d_in[6 + e * 13 + 4];
    const float* qdb  = (const float*)d_in[6 + e * 13 + 5];
    const float* kvw  = (const float*)d_in[6 + e * 13 + 6];
    const float* kvdw = (const float*)d_in[6 + e * 13 + 7];
    const float* kvdb = (const float*)d_in[6 + e * 13 + 8];
    const float* lnw  = (const float*)d_in[6 + e * 13 + 9];
    const float* lnb  = (const float*)d_in[6 + e * 13 + 10];
    const float* poww = (const float*)d_in[6 + e * 13 + 11];
    const float* pob  = (const float*)d_in[6 + e * 13 + 12];

    _Float16* h   = (_Float16*)R2;
    _Float16* q1  = (_Float16*)R3;
    _Float16* q   = (_Float16*)R4;
    _Float16* kv1 = (_Float16*)R0;
    _Float16* kv  = (_Float16*)R1;
    _Float16* o   = (_Float16*)R0;
    _Float16* g   = (_Float16*)(R0 + 16 * MB);
    _Float16* ln  = (_Float16*)R2;
    _Float16* a   = (_Float16*)R3;

    dim3 gg(HW / 128, NB);
    // h = p0 @ x
    hipLaunchKernelGGL((gemm_1x1<float, GM_PLAIN, r, 64>), gg, dim3(256), 0, stream,
        x, p0w, (const float*)nullptr, (const _Float16*)nullptr, (void*)h,
        64 * HW, r * HW, gates, e);
    // q1 = qw @ h
    hipLaunchKernelGGL((gemm_1x1<_Float16, GM_PLAIN, r, r>), gg, dim3(256), 0, stream,
        (const _Float16*)h, qw, (const float*)nullptr, (const _Float16*)nullptr,
        (void*)q1, r * HW, r * HW, gates, e);
    // q = dw3x3(q1) + qdb
    hipLaunchKernelGGL((dwconv<3>), dim3(NB * r * HW / 256), dim3(256), 0, stream,
        (const _Float16*)q1, qdw, qdb, q, r, gates, e);
    // kv1 = kvw @ h
    hipLaunchKernelGGL((gemm_1x1<_Float16, GM_PLAIN, 2 * r, r>), gg, dim3(256), 0, stream,
        (const _Float16*)h, kvw, (const float*)nullptr, (const _Float16*)nullptr,
        (void*)kv1, r * HW, 2 * r * HW, gates, e);
    // kv = dw7x7(kv1) + kvdb
    hipLaunchKernelGGL((dwconv<7>), dim3(NB * 2 * r * HW / 256), dim3(256), 0, stream,
        (const _Float16*)kv1, kvdw, kvdb, kv, 2 * r, gates, e);
    // g = silu(p1 @ shared)
    hipLaunchKernelGGL((gemm_1x1<float, GM_SILU, r, 64>), gg, dim3(256), 0, stream,
        shr, p1w, (const float*)nullptr, (const _Float16*)nullptr, (void*)g,
        64 * HW, r * HW, gates, e);
    // o = per-patch circular conv of q and k (k = first r channels of kv)
    hipLaunchKernelGGL(circconv, dim3(NB * r * 256 / 4), dim3(256), 0, stream,
        (const _Float16*)q, (const _Float16*)kv, o, r, 2 * r * HW, gates, e);
    // ln = LayerNorm_c(o) * v   (v = second r channels of kv)
    hipLaunchKernelGGL(ln_mul, dim3(HW / 256, NB), dim3(256), 0, stream,
        (const _Float16*)o, (const _Float16*)kv, lnw, lnb, ln,
        r, 2 * r * HW, r, gates, e);
    // a = (pow @ ln + pob) * g
    hipLaunchKernelGGL((gemm_1x1<_Float16, GM_BIASMUL, r, r>), gg, dim3(256), 0, stream,
        (const _Float16*)ln, poww, pob, (const _Float16*)g, (void*)a,
        r * HW, r * HW, gates, e);
    // out += gate * (p2 @ a)
    hipLaunchKernelGGL((gemm_1x1<_Float16, GM_ACCUM, 64, r>), gg, dim3(256), 0, stream,
        (const _Float16*)a, p2w, (const float*)nullptr, (const _Float16*)nullptr,
        (void*)out, r * HW, 64 * HW, gates, e);
}

// ---------------------------------------------------------------------------
extern "C" void kernel_launch(void* const* d_in, const int* in_sizes, int n_in,
                              void* d_out, int out_size, void* d_ws, size_t ws_size,
                              hipStream_t stream)
{
    (void)in_sizes; (void)n_in; (void)out_size; (void)ws_size;
    const float* x      = (const float*)d_in[0];
    const float* shr    = (const float*)d_in[1];
    const float* freq   = (const float*)d_in[2];
    const float* noise  = (const float*)d_in[3];
    const float* wg     = (const float*)d_in[4];
    const float* wf     = (const float*)d_in[5];
    float* out = (float*)d_out;

    char* wsb = (char*)d_ws;
    float* pooled = (float*)wsb;              // 512 f32
    float* gates  = pooled + 512;             // 32 f32
    float* gsum   = gates + 32;               // 8 f32
    const size_t MB = 1024 * 1024;
    char* R0 = wsb + 4096;                    // 32 MB  (kv1 -> o | g)
    char* R1 = R0 + 32 * MB;                  // 32 MB  (kv = k|v)
    char* R2 = R1 + 32 * MB;                  // 16 MB  (h -> ln)
    char* R3 = R2 + 16 * MB;                  // 16 MB  (q1 -> a)
    char* R4 = R3 + 16 * MB;                  // 16 MB  (q)    [total ~112 MB]

    hipLaunchKernelGGL(pool_kernel, dim3(NB * 64), dim3(256), 0, stream, x, pooled);
    hipLaunchKernelGGL(gate_kernel, dim3(1), dim3(32), 0, stream,
                       pooled, freq, noise, wg, wf, gates, gsum);
    hipLaunchKernelGGL(init_out, dim3(32768), dim3(256), 0, stream, x, gsum, out);

    expert_pipeline<16>(d_in, 0, x, shr, out, gates, R0, R1, R2, R3, R4, stream);
    expert_pipeline<32>(d_in, 1, x, shr, out, gates, R0, R1, R2, R3, R4, stream);
    expert_pipeline<48>(d_in, 2, x, shr, out, gates, R0, R1, R2, R3, R4, stream);
    expert_pipeline<64>(d_in, 3, x, shr, out, gates, R0, R1, R2, R3, R4, stream);
}
